// MultiHeadAttention_64132451663993
// MI455X (gfx1250) — compile-verified, matrix operand order checked by reading, not run on hardware
//
#include <hip/hip_runtime.h>
#include <hip/hip_bf16.h>

// ---------------------------------------------------------------------------
// Multi-head attention (B=2, S=2048, D=1024, H=16, Dh=64), causal, fp32 I/O.
// bf16 WMMA (v_wmma_f32_16x16x32_bf16) for all GEMM stages.
//  - projection GEMMs: 2x2 register-blocked (32x32 per wave) for 2x fragment reuse
//  - flash attention: 32-wide key tiles so the P*V WMMA runs dense (full K=32)
// ---------------------------------------------------------------------------

#define B_   2
#define S_   2048
#define D_   1024
#define H_   16
#define DH_  64

typedef __attribute__((ext_vector_type(16))) __bf16 v16bf;
typedef __attribute__((ext_vector_type(8)))  __bf16 v8bf;
typedef __attribute__((ext_vector_type(8)))  float  v8f;

__device__ __forceinline__ __bf16 f2bf(float f) {
    unsigned u = __builtin_bit_cast(unsigned, f);
    unsigned r = (u + 0x7FFFu + ((u >> 16) & 1u)) >> 16;
    unsigned short s = (unsigned short)r;
    return __builtin_bit_cast(__bf16, s);
}

__device__ __forceinline__ v8f wmma_bf16(v16bf a, v16bf b, v8f c) {
    return __builtin_amdgcn_wmma_f32_16x16x32_bf16(
        false, a, false, b, (short)0, c, false, false);
}

// 16-bit A/B fragment for v_wmma_*_16x16x32: per-lane 8 contiguous bf16 at p
// (K = koff..koff+7) and 8 contiguous at p+16 (K = koff+16..koff+23); caller
// folds koff = (lane>=16)*8 into p.
__device__ __forceinline__ v16bf load_frag(const __bf16* p) {
    v8bf lo = *(const v8bf*)(p);
    v8bf hi = *(const v8bf*)(p + 16);
    v16bf r;
#pragma unroll
    for (int i = 0; i < 8; ++i) { r[i] = lo[i]; r[i + 8] = hi[i]; }
    return r;
}

// ---------------------------------------------------------------------------
// Stage 0: conversions
// ---------------------------------------------------------------------------
__global__ void cvt_f32_to_bf16(const float* __restrict__ in,
                                __bf16* __restrict__ out, int n) {
    int i = blockIdx.x * blockDim.x + threadIdx.x;
    if (i < n) out[i] = f2bf(in[i]);
}

// Wt[n*K + k] = W[k*N + n]
__global__ void cvt_transpose(const float* __restrict__ W,
                              __bf16* __restrict__ Wt, int K, int N) {
    int n = blockIdx.x * blockDim.x + threadIdx.x;
    int k = blockIdx.y;
    if (n < N) Wt[(size_t)n * K + k] = f2bf(W[(size_t)k * N + n]);
}

// ---------------------------------------------------------------------------
// Stage 1: fused Q/K/V projection GEMM, 2x2 register-blocked.
//   grid = (S/64, D/128, B*3), block = 256 = 8 waves arranged 2(M) x 4(N);
//   each wave computes a 32x32 tile: 4 WMMAs per 4 fragment loads.
//   Q,K stored [B,H,S,Dh]; V stored transposed [B,H,Dh,S].
// ---------------------------------------------------------------------------
__global__ __launch_bounds__(256) void qkv_gemm(
    const __bf16* __restrict__ xb,
    const __bf16* __restrict__ WqT, const __bf16* __restrict__ WkT,
    const __bf16* __restrict__ WvT,
    __bf16* __restrict__ Qb, __bf16* __restrict__ Kb, __bf16* __restrict__ Vt) {

    const int lane = threadIdx.x & 31;
    const int wave = threadIdx.x >> 5;
    const int wm   = wave >> 2;         // 0..1
    const int wn   = wave & 3;          // 0..3
    const int mBase = blockIdx.x * 64 + wm * 32;
    const int nBase = blockIdx.y * 128 + wn * 32;
    const int b     = blockIdx.z / 3;
    const int which = blockIdx.z % 3;
    const int koff  = (lane >> 4) * 8;
    const int ln16  = lane & 15;

    const __bf16* Wt = (which == 0) ? WqT : (which == 1) ? WkT : WvT;

    const __bf16* aRow = xb + ((size_t)(b * S_ + mBase + ln16)) * D_ + koff;
    const __bf16* bRow = Wt + ((size_t)(nBase + ln16)) * D_ + koff;

    v8f c00 = {}, c01 = {}, c10 = {}, c11 = {};
#pragma unroll 2
    for (int k0 = 0; k0 < D_; k0 += 32) {
        __builtin_prefetch(aRow + k0 + 256, 0, 1);
        __builtin_prefetch(bRow + k0 + 256, 0, 1);
        v16bf a0 = load_frag(aRow + k0);
        v16bf a1 = load_frag(aRow + (size_t)16 * D_ + k0);
        v16bf w0 = load_frag(bRow + k0);
        v16bf w1 = load_frag(bRow + (size_t)16 * D_ + k0);
        c00 = wmma_bf16(a0, w0, c00);
        c01 = wmma_bf16(a0, w1, c01);
        c10 = wmma_bf16(a1, w0, c10);
        c11 = wmma_bf16(a1, w1, c11);
    }

    // C layout: VGPR r -> row (lane<16 ? r : r+8), column lane&15.
#pragma unroll
    for (int i = 0; i < 2; ++i) {
#pragma unroll
        for (int j = 0; j < 2; ++j) {
            const v8f& c = (i == 0) ? (j == 0 ? c00 : c01)
                                    : (j == 0 ? c10 : c11);
            const int ncol = nBase + j * 16 + ln16;
            const int h = ncol >> 6;
            const int d = ncol & 63;
#pragma unroll
            for (int r = 0; r < 8; ++r) {
                int row = mBase + i * 16 + ((lane < 16) ? r : r + 8);
                __bf16 v = f2bf(c[r]);
                if (which == 0) {
                    Qb[((size_t)(b * H_ + h) * S_ + row) * DH_ + d] = v;
                } else if (which == 1) {
                    Kb[((size_t)(b * H_ + h) * S_ + row) * DH_ + d] = v;
                } else {
                    Vt[((size_t)(b * H_ + h) * DH_ + d) * S_ + row] = v;
                }
            }
        }
    }
}

// ---------------------------------------------------------------------------
// Stage 2: causal flash attention, one wave per 16-query tile per (b,h),
// 32-wide key tiles (dense K=32 P*V WMMAs).
//   grid = (S/16/8, H, B), block = 256 (8 independent waves).
// ---------------------------------------------------------------------------
__global__ __launch_bounds__(256) void attn_flash(
    const __bf16* __restrict__ Qb, const __bf16* __restrict__ Kb,
    const __bf16* __restrict__ Vt, __bf16* __restrict__ ctx) {

    __shared__ float ptile[8][16 * 33];   // per-wave 16x32 P staging (+pad)

    const int lane = threadIdx.x & 31;
    const int wave = threadIdx.x >> 5;
    const int mt   = blockIdx.x * 8 + wave;
    const int h    = blockIdx.y;
    const int b    = blockIdx.z;
    const int m0   = mt * 16;
    const int koff = (lane >> 4) * 8;
    const int ln16 = lane & 15;

    const __bf16* Qh = Qb + (size_t)(b * H_ + h) * S_ * DH_;
    const __bf16* Kh = Kb + (size_t)(b * H_ + h) * S_ * DH_;
    const __bf16* Vh = Vt + (size_t)(b * H_ + h) * DH_ * S_;

    // Q fragments (K-dim 0..31 and 32..63), resident all loop
    const __bf16* qrow = Qh + (size_t)(m0 + ln16) * DH_ + koff;
    const v16bf qa0 = load_frag(qrow);
    const v16bf qa1 = load_frag(qrow + 32);

    v8f acc0 = {}, acc1 = {}, acc2 = {}, acc3 = {};
    float rm[8], rl[8];
#pragma unroll
    for (int r = 0; r < 8; ++r) { rm[r] = -3.0e38f; rl[r] = 0.0f; }

    float* myP = ptile[wave];

    // 32-wide key tiles; ragged diagonal handled by the mask (exp -> 0),
    // and j0 <= m0 <= S-16 keeps all loads in bounds.
    for (int j0 = 0; j0 <= m0; j0 += 32) {
        // two 16-wide score tiles: key rows j0+n and j0+16+n
        const __bf16* krow0 = Kh + (size_t)(j0 + ln16) * DH_ + koff;
        const __bf16* krow1 = krow0 + (size_t)16 * DH_;
        v8f s0 = {}, s1 = {};
        s0 = wmma_bf16(qa0, load_frag(krow0), s0);
        s0 = wmma_bf16(qa1, load_frag(krow0 + 32), s0);
        s1 = wmma_bf16(qa0, load_frag(krow1), s1);
        s1 = wmma_bf16(qa1, load_frag(krow1 + 32), s1);

        const int col0 = j0 + ln16;
        const int col1 = col0 + 16;
#pragma unroll
        for (int r = 0; r < 8; ++r) {
            int row = m0 + ((lane < 16) ? r : r + 8);
            float v0 = s0[r] * 0.125f;               // 1/sqrt(64)
            float v1 = s1[r] * 0.125f;
            if (col0 > row) v0 = -3.0e38f;           // causal mask
            if (col1 > row) v1 = -3.0e38f;
            s0[r] = v0; s1[r] = v1;
        }

        // online softmax per row (32 columns: lane pair + 16-lane reduce)
#pragma unroll
        for (int r = 0; r < 8; ++r) {
            float v = fmaxf(s0[r], s1[r]);
            v = fmaxf(v, __shfl_xor(v, 8, 32));
            v = fmaxf(v, __shfl_xor(v, 4, 32));
            v = fmaxf(v, __shfl_xor(v, 2, 32));
            v = fmaxf(v, __shfl_xor(v, 1, 32));
            float mnew = fmaxf(rm[r], v);
            float corr = __expf(rm[r] - mnew);
            float p0   = __expf(s0[r] - mnew);
            float p1   = __expf(s1[r] - mnew);
            float ps   = p0 + p1;
            ps += __shfl_xor(ps, 8, 32);
            ps += __shfl_xor(ps, 4, 32);
            ps += __shfl_xor(ps, 2, 32);
            ps += __shfl_xor(ps, 1, 32);
            rl[r] = rl[r] * corr + ps;
            rm[r] = mnew;
            s0[r] = p0; s1[r] = p1;
            acc0[r] *= corr; acc1[r] *= corr;
            acc2[r] *= corr; acc3[r] *= corr;
        }

        // stage 16x32 P through LDS: C layout -> A layout (same-wave DS order)
#pragma unroll
        for (int r = 0; r < 8; ++r) {
            int row = (lane < 16) ? r : r + 8;
            myP[row * 33 + ln16]      = s0[r];
            myP[row * 33 + 16 + ln16] = s1[r];
        }

        // full-K A fragment: elems 0..7 -> P[m][koff+e], 8..15 -> P[m][koff+16+e]
        v16bf pa;
#pragma unroll
        for (int e = 0; e < 8; ++e) {
            pa[e]     = f2bf(myP[ln16 * 33 + koff + e]);
            pa[e + 8] = f2bf(myP[ln16 * 33 + koff + 16 + e]);
        }

        // O += P V : 4 n-tiles of 16 over Dh=64; V is [Dh,S] so the dense
        // K=32 B fragment is exactly load_frag (contiguous in s).
#pragma unroll
        for (int t = 0; t < 4; ++t) {
            const __bf16* vptr = Vh + (size_t)(t * 16 + ln16) * S_ + j0 + koff;
            v16bf vb = load_frag(vptr);
            if      (t == 0) acc0 = wmma_bf16(pa, vb, acc0);
            else if (t == 1) acc1 = wmma_bf16(pa, vb, acc1);
            else if (t == 2) acc2 = wmma_bf16(pa, vb, acc2);
            else             acc3 = wmma_bf16(pa, vb, acc3);
        }
    }

    // normalize and store ctx [B,S,D] bf16
#pragma unroll
    for (int r = 0; r < 8; ++r) {
        float inv = 1.0f / rl[r];
        int row = m0 + ((lane < 16) ? r : r + 8);
        size_t base = ((size_t)(b * S_ + row)) * D_ + h * DH_ + ln16;
        ctx[base + 0 ] = f2bf(acc0[r] * inv);
        ctx[base + 16] = f2bf(acc1[r] * inv);
        ctx[base + 32] = f2bf(acc2[r] * inv);
        ctx[base + 48] = f2bf(acc3[r] * inv);
    }
}

// ---------------------------------------------------------------------------
// Stage 3: output projection  out = ctx @ Wo + bo  (fp32 out), 2x2 blocked.
//   grid = (B*S/64, D/128), block = 256
// ---------------------------------------------------------------------------
__global__ __launch_bounds__(256) void out_proj(
    const __bf16* __restrict__ ctx, const __bf16* __restrict__ WoT,
    const float* __restrict__ bo, float* __restrict__ out) {

    const int lane = threadIdx.x & 31;
    const int wave = threadIdx.x >> 5;
    const int wm   = wave >> 2;
    const int wn   = wave & 3;
    const int mBase = blockIdx.x * 64 + wm * 32;
    const int nBase = blockIdx.y * 128 + wn * 32;
    const int koff  = (lane >> 4) * 8;
    const int ln16  = lane & 15;

    const __bf16* aRow = ctx + (size_t)(mBase + ln16) * D_ + koff;
    const __bf16* bRow = WoT + (size_t)(nBase + ln16) * D_ + koff;

    v8f c00 = {}, c01 = {}, c10 = {}, c11 = {};
#pragma unroll 2
    for (int k0 = 0; k0 < D_; k0 += 32) {
        __builtin_prefetch(aRow + k0 + 256, 0, 1);
        __builtin_prefetch(bRow + k0 + 256, 0, 1);
        v16bf a0 = load_frag(aRow + k0);
        v16bf a1 = load_frag(aRow + (size_t)16 * D_ + k0);
        v16bf w0 = load_frag(bRow + k0);
        v16bf w1 = load_frag(bRow + (size_t)16 * D_ + k0);
        c00 = wmma_bf16(a0, w0, c00);
        c01 = wmma_bf16(a0, w1, c01);
        c10 = wmma_bf16(a1, w0, c10);
        c11 = wmma_bf16(a1, w1, c11);
    }

#pragma unroll
    for (int i = 0; i < 2; ++i) {
#pragma unroll
        for (int j = 0; j < 2; ++j) {
            const v8f& c = (i == 0) ? (j == 0 ? c00 : c01)
                                    : (j == 0 ? c10 : c11);
            const int ncol = nBase + j * 16 + ln16;
            const float bias = bo[ncol];
#pragma unroll
            for (int r = 0; r < 8; ++r) {
                int row = mBase + i * 16 + ((lane < 16) ? r : r + 8);
                out[(size_t)row * D_ + ncol] = c[r] + bias;
            }
        }
    }
}

// ---------------------------------------------------------------------------
// Launch
// ---------------------------------------------------------------------------
extern "C" void kernel_launch(void* const* d_in, const int* in_sizes, int n_in,
                              void* d_out, int out_size, void* d_ws, size_t ws_size,
                              hipStream_t stream) {
    (void)in_sizes; (void)n_in; (void)out_size; (void)ws_size;

    const float* x  = (const float*)d_in[0];
    const float* Wq = (const float*)d_in[1];
    const float* Wk = (const float*)d_in[2];
    const float* Wv = (const float*)d_in[3];
    const float* Wo = (const float*)d_in[4];
    const float* bo = (const float*)d_in[5];
    float* out = (float*)d_out;

    char* ws = (char*)d_ws;
    const size_t MB = 1ull << 20;
    __bf16* xb  = (__bf16*)(ws);               //  8 MiB: x in bf16
    __bf16* WqT = (__bf16*)(ws +  8 * MB);     //  2 MiB each, transposed
    __bf16* WkT = (__bf16*)(ws + 10 * MB);
    __bf16* WvT = (__bf16*)(ws + 12 * MB);
    __bf16* WoT = (__bf16*)(ws + 14 * MB);
    __bf16* Qb  = (__bf16*)(ws + 16 * MB);     //  8 MiB [B,H,S,Dh]
    __bf16* Kb  = (__bf16*)(ws + 24 * MB);     //  8 MiB [B,H,S,Dh]
    __bf16* Vt  = (__bf16*)(ws + 32 * MB);     //  8 MiB [B,H,Dh,S]
    __bf16* ctx = (__bf16*)(ws + 40 * MB);     //  8 MiB [B,S,D]

    const int nx = B_ * S_ * D_;
    cvt_f32_to_bf16<<<dim3((nx + 255) / 256), dim3(256), 0, stream>>>(x, xb, nx);
    cvt_transpose<<<dim3(D_ / 256, D_), dim3(256), 0, stream>>>(Wq, WqT, D_, D_);
    cvt_transpose<<<dim3(D_ / 256, D_), dim3(256), 0, stream>>>(Wk, WkT, D_, D_);
    cvt_transpose<<<dim3(D_ / 256, D_), dim3(256), 0, stream>>>(Wv, WvT, D_, D_);
    cvt_transpose<<<dim3(D_ / 256, D_), dim3(256), 0, stream>>>(Wo, WoT, D_, D_);

    qkv_gemm<<<dim3(S_ / 64, D_ / 128, B_ * 3), dim3(256), 0, stream>>>(
        xb, WqT, WkT, WvT, Qb, Kb, Vt);

    attn_flash<<<dim3(S_ / 16 / 8, H_, B_), dim3(256), 0, stream>>>(
        Qb, Kb, Vt, ctx);

    out_proj<<<dim3(B_ * S_ / 64, D_ / 128), dim3(256), 0, stream>>>(
        ctx, WoT, bo, out);
}
